// QLSTM_16862041604454
// MI455X (gfx1250) — matrix operand order repaired and checked by reading, not compiled, not run on hardware
//
#include <hip/hip_runtime.h>

// ---------------------------------------------------------------------------
// Quaternion LSTM for MI455X (gfx1250), wave32 + WMMA bf16 (f32 accumulate).
// T=500, B=32, D=H=1024.
// K-loops are software-pipelined with __builtin_amdgcn_sched_barrier fences so
// the backend cannot collapse the operand double-buffer: each v_wmma waits only
// on loads issued >=1 (gemm_pre) / >=4 (step) iterations earlier.
// ---------------------------------------------------------------------------

typedef __bf16  bf16x16 __attribute__((ext_vector_type(16)));
typedef float   v8f     __attribute__((ext_vector_type(8)));

#if __has_builtin(__builtin_amdgcn_sched_barrier)
#define SCHED_FENCE() __builtin_amdgcn_sched_barrier(0)
#else
#define SCHED_FENCE()
#endif

static constexpr int T  = 500;
static constexpr int Bn = 32;
static constexpr int D  = 1024;
static constexpr int H  = 1024;

static constexpr int TB   = T * Bn;       // 16000 rows of x
static constexpr int RT   = TB / 16;      // 1000 row tiles (never cross t: 32%16==0)
static constexpr int KT_D = D / 32;       // 32 K-tiles for input GEMM
static constexpr int KT_H = H / 32;       // 32 K-tiles for recurrent GEMM
static constexpr int NT_H = H / 16;       // 64 N-tiles per gate

// quaternion Hamilton cat: comp/sign indexed [colBlock][rowBlock]
__device__ __constant__ int   QCOMP[4][4] = {{0,1,2,3},{1,0,3,2},{2,3,0,1},{3,2,1,0}};
__device__ __constant__ float QSIGN[4][4] = {{1.f,-1.f,-1.f,-1.f},
                                             {1.f, 1.f,-1.f, 1.f},
                                             {1.f, 1.f, 1.f,-1.f},
                                             {1.f,-1.f, 1.f, 1.f}};

__device__ __forceinline__ unsigned short f2bf(float f) {
    unsigned int u = __float_as_uint(f);
    unsigned int r = u + 0x7FFFu + ((u >> 16) & 1u);   // round-to-nearest-even
    return (unsigned short)(r >> 16);
}
__device__ __forceinline__ float sigm(float x) { return 1.f / (1.f + __expf(-x)); }

// A-operand 16-bit layout: lane = hi*16 + m (row), vec elem i -> K
__device__ __forceinline__ int kmapA(int i, int hi) {
    return ((i >> 3) << 4) + (i & 7) + (hi << 3);
}

// ---------------------------------------------------------------------------
// Pack x (f32 [T,B,D]) into bf16 A-operand tiles: [RT][KT_D][32 lanes][16]
// ---------------------------------------------------------------------------
__global__ void qlstm_pack_x(const float* __restrict__ x, unsigned short* __restrict__ xp) {
    int idx = blockIdx.x * 256 + threadIdx.x;
    if (idx >= RT * KT_D * 512) return;
    int e    = idx & 511;
    int tile = idx >> 9;
    int kt   = tile % KT_D;
    int rt   = tile / KT_D;
    int lane = e >> 4, i = e & 15;
    int hi   = lane >> 4, m = lane & 15;
    int K    = kt * 32 + kmapA(i, hi);
    int R    = rt * 16 + m;                       // R = t*32 + b
    xp[idx]  = f2bf(x[(size_t)R * D + K]);
}

// ---------------------------------------------------------------------------
// Pack quaternion weights into bf16 B-operand tiles: [4 g][NT][KT][32 lanes][16]
// B 32x16 layout: lane = hi*16 + n, elem i -> K = 16*(i/8) + (i%8) + 8*hi.
// Hamilton cat folded in: w[(g,comp,d',h')] * sign.
// ---------------------------------------------------------------------------
__global__ void qlstm_pack_w(const float* __restrict__ wx, const float* __restrict__ wh,
                             unsigned short* __restrict__ wxp, unsigned short* __restrict__ whp) {
    const int PER = 4 * NT_H * KT_D * 512;        // 4,194,304
    int idx = blockIdx.x * 256 + threadIdx.x;
    if (idx >= 2 * PER) return;
    const float* src = (idx < PER) ? wx : wh;
    unsigned short* dst = (idx < PER) ? wxp : whp;
    int j = (idx < PER) ? idx : idx - PER;

    int e    = j & 511;
    int tile = j >> 9;
    int kt   = tile & 31;
    int nt   = (tile >> 5) & 63;
    int g    = tile >> 11;
    int lane = e >> 4, i = e & 15;
    int hi   = lane >> 4, nl = lane & 15;
    int K    = kt * 32 + kmapA(i, hi);            // row index into [D|H]
    int n    = nt * 16 + nl;                      // col index into H
    int rb   = K >> 8, dq = K & 255;              // D/4 = H/4 = 256
    int cb   = n >> 8, hq = n & 255;
    int comp = QCOMP[cb][rb];
    float v  = QSIGN[cb][rb] * src[(((size_t)g * 4 + comp) * 256 + dq) * 256 + hq];
    dst[j]   = f2bf(v);
}

// ---------------------------------------------------------------------------
// Zero c state and both h A-operand buffers.
// ---------------------------------------------------------------------------
__global__ void qlstm_init(float* __restrict__ c, unsigned int* __restrict__ hb) {
    int idx = blockIdx.x * 256 + threadIdx.x;
    if (idx < Bn * H) c[idx] = 0.f;                        // 32768 floats
    if (idx < (2 * 2 * KT_H * 512 * 2) / 4) hb[idx] = 0u;  // 32768 dwords
}

// ---------------------------------------------------------------------------
// Big input GEMM: pre[t][g][b][h] = x @ wxc + bx. One wave per 16x64 output
// strip (1 A-tile reused across 4 B-tiles). 64000 waves = 16000 blocks x 128.
// Depth-1 pipeline with sched fences: loads(kt+1) || wmma(kt).
// ---------------------------------------------------------------------------
__global__ void __launch_bounds__(128)
qlstm_gemm_pre(const unsigned short* __restrict__ xp,
               const unsigned short* __restrict__ wxp,
               const float* __restrict__ bx,
               float* __restrict__ pre) {
    int wave = blockIdx.x * 4 + (threadIdx.x >> 5);
    int lane = threadIdx.x & 31;
    int cg = wave & 15;                 // column group (4 N-tiles)
    int g  = (wave >> 4) & 3;
    int rt = wave >> 6;

    // per-lane operand pointers; kt stride = 32 vectors, j stride = KT_D*32 = 1024
    const bf16x16* Ab = reinterpret_cast<const bf16x16*>(xp)
                      + (((size_t)rt * KT_D) << 5) + lane;
    const bf16x16* Bb = reinterpret_cast<const bf16x16*>(wxp)
                      + (((size_t)(g * NT_H + cg * 4) * KT_D) << 5) + lane;

    v8f acc[4];
    #pragma unroll
    for (int j = 0; j < 4; ++j) acc[j] = v8f{};

    bf16x16 a_cur = Ab[0];
    bf16x16 b_cur[4];
    #pragma unroll
    for (int j = 0; j < 4; ++j) b_cur[j] = Bb[j * 1024];

    #pragma unroll 2
    for (int kt = 0; kt < KT_D - 1; ++kt) {
        // phase 1: issue next-iteration operand loads + WGP-scope prefetch ahead
        bf16x16 a_nxt = Ab[(kt + 1) << 5];
        bf16x16 b_nxt[4];
        #pragma unroll
        for (int j = 0; j < 4; ++j) b_nxt[j] = Bb[j * 1024 + ((kt + 1) << 5)];
        __builtin_prefetch((const void*)(Ab + ((kt + 8) << 5)), 0, 3);
        __builtin_prefetch((const void*)(Bb + ((kt + 8) << 5)), 0, 3);
        SCHED_FENCE();
        // phase 2: matrix math on current operands (overlaps loads above)
        #pragma unroll
        for (int j = 0; j < 4; ++j)
            acc[j] = __builtin_amdgcn_wmma_f32_16x16x32_bf16(
                         false, a_cur, false, b_cur[j], (short)0, acc[j], false, false);
        SCHED_FENCE();
        a_cur = a_nxt;
        #pragma unroll
        for (int j = 0; j < 4; ++j) b_cur[j] = b_nxt[j];
    }
    #pragma unroll
    for (int j = 0; j < 4; ++j)
        acc[j] = __builtin_amdgcn_wmma_f32_16x16x32_bf16(
                     false, a_cur, false, b_cur[j], (short)0, acc[j], false, false);

    int hi = lane >> 4, nl = lane & 15;
    #pragma unroll
    for (int j = 0; j < 4; ++j) {
        int n = (cg * 4 + j) * 16 + nl;
        float bias = bx[g * H + n];
        #pragma unroll
        for (int v = 0; v < 8; ++v) {
            int m  = v + (hi << 3);
            int R  = rt * 16 + m;
            int t  = R >> 5, bb = R & 31;
            pre[(((size_t)t * 4 + g) * Bn + bb) * H + n] = acc[j][v] + bias;
        }
    }
}

// ---------------------------------------------------------------------------
// One recurrence step. Block = 128 threads = 4 waves (one per gate), grid =
// 2 M-tiles x 64 N-tiles = 128 blocks. Depth-4 fully-unrolled WMMA pipeline
// (ring rotation is pure SSA renaming), gate fusion via LDS, pointwise c/h
// update, h re-packed bf16 (double-buffered across step launches).
// ---------------------------------------------------------------------------
__global__ void __launch_bounds__(128)
qlstm_step(const unsigned short* __restrict__ hin,
           unsigned short* __restrict__ hout,
           const unsigned short* __restrict__ whp,
           const float* __restrict__ pre_t,     // [4][32][1024]
           float* __restrict__ c,               // [32][1024]
           float* __restrict__ hs_t) {          // [32][1024]
    __shared__ float lds[4 * 256];

    int g    = threadIdx.x >> 5;
    int lane = threadIdx.x & 31;
    int mt   = blockIdx.x >> 6;      // 0..1
    int nt   = blockIdx.x & 63;      // 0..63

    const bf16x16* Ab = reinterpret_cast<const bf16x16*>(hin)
                      + ((mt * KT_H) << 5) + lane;
    const bf16x16* Bb = reinterpret_cast<const bf16x16*>(whp)
                      + (((size_t)(g * NT_H + nt) * KT_H) << 5) + lane;

    v8f acc = v8f{};
    bf16x16 aw[4], bw[4];
    #pragma unroll
    for (int d = 0; d < 4; ++d) { aw[d] = Ab[d << 5]; bw[d] = Bb[d << 5]; }

    #pragma unroll
    for (int kt = 0; kt < KT_H - 4; ++kt) {
        const int s = kt & 3;
        // issue loads 4 iterations ahead; prefetch whc stream 12 tiles ahead
        bf16x16 a_n = Ab[(kt + 4) << 5];
        bf16x16 b_n = Bb[(kt + 4) << 5];
        __builtin_prefetch((const void*)(Bb + ((kt + 12) << 5)), 0, 3);
        SCHED_FENCE();
        acc = __builtin_amdgcn_wmma_f32_16x16x32_bf16(
                  false, aw[s], false, bw[s], (short)0, acc, false, false);
        SCHED_FENCE();
        aw[s] = a_n; bw[s] = b_n;          // SSA rename under full unroll
    }
    #pragma unroll
    for (int kt = KT_H - 4; kt < KT_H; ++kt)
        acc = __builtin_amdgcn_wmma_f32_16x16x32_bf16(
                  false, aw[kt & 3], false, bw[kt & 3], (short)0, acc, false, false);

    int hi = lane >> 4, nl = lane & 15;
    #pragma unroll
    for (int v = 0; v < 8; ++v) {
        int m  = v + (hi << 3);
        int bb = (mt << 4) + m;
        int n  = (nt << 4) + nl;
        float val = acc[v] + pre_t[((g * Bn + bb) << 10) + n];
        lds[g * 256 + m * 16 + nl] = val;
    }
    __syncthreads();

    // pointwise: 256 elements over 128 threads
    #pragma unroll
    for (int r = 0; r < 2; ++r) {
        int e  = threadIdx.x + r * 128;
        int m  = e >> 4, nl2 = e & 15;
        int bb = (mt << 4) + m;
        int n  = (nt << 4) + nl2;
        float ft = sigm(lds[       e]);
        float it = sigm(lds[256  + e]);
        float ot = sigm(lds[512  + e]);
        float gt = tanhf(lds[768 + e]);
        size_t ci = ((size_t)bb << 10) + n;
        float cn = it * gt + ft * c[ci];
        float h  = ot * tanhf(cn);
        c[ci]    = cn;
        hs_t[ci] = h;
        // repack h into A-operand layout for next step (row bb, K = n)
        int kt2 = n >> 5, kk = n & 31;
        int hi2 = (kk >> 3) & 1;
        int i2  = ((kk >> 4) << 3) + (kk & 7);
        int ln2 = (hi2 << 4) + m;
        hout[((size_t)(mt * KT_H + kt2) << 9) + (ln2 << 4) + i2] = f2bf(h);
    }
}

// ---------------------------------------------------------------------------
// Host-side orchestration
// ---------------------------------------------------------------------------
extern "C" void kernel_launch(void* const* d_in, const int* in_sizes, int n_in,
                              void* d_out, int out_size, void* d_ws, size_t ws_size,
                              hipStream_t stream) {
    (void)in_sizes; (void)n_in; (void)out_size; (void)ws_size;

    const float* x  = (const float*)d_in[0];   // [T,B,D]
    const float* wx = (const float*)d_in[1];   // [4,4,256,256]
    const float* wh = (const float*)d_in[2];   // [4,4,256,256]
    const float* bx = (const float*)d_in[3];   // [4,H]
    float* hs       = (float*)d_out;           // [T,B,H]

    char* ws = (char*)d_ws;
    const size_t XP_ELEMS   = (size_t)RT * KT_D * 512;        // 16,384,000 bf16
    const size_t W_ELEMS    = (size_t)4 * NT_H * KT_D * 512;  //  4,194,304 bf16 each
    const size_t PRE_ELEMS  = (size_t)T * 4 * Bn * H;         // 65,536,000 f32
    const size_t HBUF_ELEMS = (size_t)2 * KT_H * 512;         //     32,768 bf16 each

    unsigned short* xp  = (unsigned short*)ws;                 ws += XP_ELEMS * 2;
    unsigned short* wxp = (unsigned short*)ws;                 ws += W_ELEMS * 2;
    unsigned short* whp = (unsigned short*)ws;                 ws += W_ELEMS * 2;
    float*          pre = (float*)ws;                          ws += PRE_ELEMS * 4;
    float*          c   = (float*)ws;                          ws += (size_t)Bn * H * 4;
    unsigned short* hb  = (unsigned short*)ws;                 // 2 buffers, 32768 each

    // 1. pack inputs/weights to bf16 WMMA-native layout
    qlstm_pack_x<<<(XP_ELEMS + 255) / 256, 256, 0, stream>>>(x, xp);
    qlstm_pack_w<<<(2 * W_ELEMS + 255) / 256, 256, 0, stream>>>(wx, wh, wxp, whp);
    qlstm_init<<<256, 256, 0, stream>>>(c, (unsigned int*)hb);

    // 2. big parallel GEMM: pre = x @ wxc + bx
    qlstm_gemm_pre<<<16000, 128, 0, stream>>>(xp, wxp, bx, pre);

    // 3. sequential recurrence, double-buffered h operand
    for (int t = 0; t < T; ++t) {
        const unsigned short* hin = hb + (size_t)(t & 1) * HBUF_ELEMS;
        unsigned short* hout      = hb + (size_t)((t + 1) & 1) * HBUF_ELEMS;
        qlstm_step<<<128, 128, 0, stream>>>(
            hin, hout, whp,
            pre + (size_t)t * 4 * Bn * H,
            c,
            hs + (size_t)t * Bn * H);
    }
}
